// MultiHeadAttention_73332271612736
// MI455X (gfx1250) — matrix operand order repaired and checked by reading, not compiled, not run on hardware
//
#include <hip/hip_runtime.h>
#include <hip/hip_bf16.h>

#define BB 4
#define SS 1024
#define EE 512
#define AA 64
#define HH 8

typedef __bf16 bf16_t;
typedef __attribute__((ext_vector_type(16))) __bf16 v16bf;
typedef __attribute__((ext_vector_type(8)))  __bf16 bf16x8;
typedef __attribute__((ext_vector_type(8)))  float  v8f;

#define WMMA_BF16(a, b, c) \
  __builtin_amdgcn_wmma_f32_16x16x32_bf16(false, (a), false, (b), (short)0, (c), false, false)

#define SCHED_FENCE() __builtin_amdgcn_sched_barrier(0)

// ---------------------------------------------------------------------------
// Fragment gathers per CDNA5 ISA 7.12.2 (wave32), K-contiguous memory so each
// fragment is exactly two 128-bit loads.
// A-matrix 16x32 bf16: lane holds row r = lane&15;
//   elems 0..7  -> K = (lane>>4)*8 + e        (one b128)
//   elems 8..15 -> K = 16 + (lane>>4)*8 + e-8 (one b128)
// B-matrix 32x16 bf16: lane holds col n = lane&15;
//   elems 0..7  -> K = (lane>>4)*8 + e        (one b128)
//   elems 8..15 -> K = 16 + (lane>>4)*8 + e-8 (one b128)
// C/D 16x16 f32: lane holds col n = lane&15, rows m = (lane>>4)*8 + v.
// ---------------------------------------------------------------------------
__device__ __forceinline__ v16bf frag_a_k(const bf16_t* p, int rs) {
  const int lane = threadIdx.x & 31;
  const int base = (lane & 15) * rs + ((lane >> 4) << 3);
  union { v16bf v; bf16x8 h[2]; } u;
  u.h[0] = *(const bf16x8*)(p + base);
  u.h[1] = *(const bf16x8*)(p + base + 16);
  return u.v;
}

__device__ __forceinline__ v16bf frag_b_k(const bf16_t* p, int ns) {
  const int lane = threadIdx.x & 31;
  const int base = (lane & 15) * ns + ((lane >> 4) << 3);
  union { v16bf v; bf16x8 h[2]; } u;
  u.h[0] = *(const bf16x8*)(p + base);
  u.h[1] = *(const bf16x8*)(p + base + 16);
  return u.v;
}

// ---------------------------------------------------------------------------
// Conversions (fp32 -> bf16), with transposes for the weight operands so all
// B-fragments become K-contiguous.
// ---------------------------------------------------------------------------
__global__ void cvt_bf16x8_kernel(const float* __restrict__ in,
                                  bf16_t* __restrict__ out, int n8) {
  int i = blockIdx.x * blockDim.x + threadIdx.x;
  if (i < n8) {
    const float4 f0 = ((const float4*)in)[2 * i];
    const float4 f1 = ((const float4*)in)[2 * i + 1];
    bf16x8 o;
    o[0] = (bf16_t)f0.x; o[1] = (bf16_t)f0.y;
    o[2] = (bf16_t)f0.z; o[3] = (bf16_t)f0.w;
    o[4] = (bf16_t)f1.x; o[5] = (bf16_t)f1.y;
    o[6] = (bf16_t)f1.z; o[7] = (bf16_t)f1.w;
    ((bf16x8*)out)[i] = o;
  }
}

// wq/wk/wv: [H][E][A] f32  ->  [H][A][E] bf16
__global__ void cvt_t_hea_kernel(const float* __restrict__ in,
                                 bf16_t* __restrict__ out) {
  int i = blockIdx.x * blockDim.x + threadIdx.x;
  if (i < HH * EE * AA) {
    int a = i % AA, e = (i / AA) % EE, h = i / (AA * EE);
    out[((size_t)h * AA + a) * EE + e] = (bf16_t)in[i];
  }
}

// wo: [H*A][A] f32 -> [A][H*A] bf16
__global__ void cvt_t_wo_kernel(const float* __restrict__ in,
                                bf16_t* __restrict__ out) {
  int i = blockIdx.x * blockDim.x + threadIdx.x;
  if (i < (HH * AA) * AA) {
    int a = i % AA, k = i / AA;
    out[(size_t)a * (HH * AA) + k] = (bf16_t)in[i];
  }
}

// ---------------------------------------------------------------------------
// QKV projection. Block = 128 threads (4 waves); wave owns a 16(s) x 64(a)
// output tile (4 accumulators). No LDS; fully unrolled, double-buffered
// K-loop with a sched_barrier pinning next-iteration loads above the current
// WMMA group -> partial s_wait_loadcnt, one full load group always in flight.
// which = 0/1/2 -> Q/K/V. V stored transposed [bh][a][s].
// ---------------------------------------------------------------------------
__global__ __launch_bounds__(128) void qkv_gemm_kernel(
    const bf16_t* __restrict__ xb, const bf16_t* __restrict__ wqT,
    const bf16_t* __restrict__ wkT, const bf16_t* __restrict__ wvT,
    const float* __restrict__ bq, const float* __restrict__ bk,
    const float* __restrict__ bv, bf16_t* __restrict__ Qb,
    bf16_t* __restrict__ Kb, bf16_t* __restrict__ VbT) {
  const int tid = threadIdx.x, wave = tid >> 5, lane = tid & 31;
  const int s0 = (blockIdx.x << 6) + (wave << 4);
  const int bh = blockIdx.y, h = bh & (HH - 1), b = bh >> 3;
  const int which = blockIdx.z;

  const bf16_t* wT  = (which == 0) ? wqT : (which == 1) ? wkT : wvT;
  const float* bias = (which == 0) ? bq  : (which == 1) ? bk  : bv;

  const bf16_t* xrow = xb + ((size_t)b * SS + s0) * EE;
  const bf16_t* w0   = wT + (size_t)h * AA * EE;     // [a][e]

  v8f acc0 = {}, acc1 = {}, acc2 = {}, acc3 = {};
  v16bf a_c  = frag_a_k(xrow, EE);
  v16bf b0_c = frag_b_k(w0 + 0 * 16 * EE, EE);
  v16bf b1_c = frag_b_k(w0 + 1 * 16 * EE, EE);
  v16bf b2_c = frag_b_k(w0 + 2 * 16 * EE, EE);
  v16bf b3_c = frag_b_k(w0 + 3 * 16 * EE, EE);
#pragma unroll
  for (int it = 0; it < EE / 32; ++it) {
    v16bf a_n, b0_n, b1_n, b2_n, b3_n;
    if (it + 1 < EE / 32) {
      const int k1 = (it + 1) * 32;
      a_n  = frag_a_k(xrow + k1, EE);
      b0_n = frag_b_k(w0 + 0 * 16 * EE + k1, EE);
      b1_n = frag_b_k(w0 + 1 * 16 * EE + k1, EE);
      b2_n = frag_b_k(w0 + 2 * 16 * EE + k1, EE);
      b3_n = frag_b_k(w0 + 3 * 16 * EE + k1, EE);
    }
    SCHED_FENCE();   // keep next-iter loads above this WMMA group
    acc0 = WMMA_BF16(a_c, b0_c, acc0);
    acc1 = WMMA_BF16(a_c, b1_c, acc1);
    acc2 = WMMA_BF16(a_c, b2_c, acc2);
    acc3 = WMMA_BF16(a_c, b3_c, acc3);
    SCHED_FENCE();
    if (it + 1 < EE / 32) {
      a_c = a_n; b0_c = b0_n; b1_c = b1_n; b2_c = b2_n; b3_c = b3_n;
    }
  }

  v8f* accs[4] = {&acc0, &acc1, &acc2, &acc3};
  if (which < 2) {
    bf16_t* orow = ((which == 0) ? Qb : Kb) + ((size_t)bh * SS + s0) * AA;
#pragma unroll
    for (int tn = 0; tn < 4; ++tn) {
      const int ncol = (tn << 4) + (lane & 15);
      const float bval = bias[h * AA + ncol];
#pragma unroll
      for (int v = 0; v < 8; ++v) {
        const int m = ((lane >> 4) << 3) + v;
        orow[m * AA + ncol] = (bf16_t)((*accs[tn])[v] + bval);
      }
    }
  } else {
    bf16_t* ob = VbT + (size_t)bh * AA * SS;
#pragma unroll
    for (int tn = 0; tn < 4; ++tn) {
      const int ncol = (tn << 4) + (lane & 15);
      const float bval = bias[h * AA + ncol];
      bf16_t* oc = ob + (size_t)ncol * SS + s0 + ((lane >> 4) << 3);
#pragma unroll
      for (int v = 0; v < 8; ++v) oc[v] = (bf16_t)((*accs[tn])[v] + bval);
    }
  }
}

// ---------------------------------------------------------------------------
// Causal flash attention. Block = 256 threads (8 independent waves); wave
// owns query tile [q0, q0+16). 32 keys per step (two 16-wide score tiles) so
// P is a full K=32 WMMA A-operand. No block barriers. Double-buffered via
// alternating fragment sets; load order (V this step, K next step) + sched
// fences give partial waits: QK^T waits at loadcnt<=16, PV at loadcnt<=8,
// with softmax + P LDS round-trip fully under outstanding loads.
// ---------------------------------------------------------------------------
__global__ __launch_bounds__(256) void attn_kernel(
    const bf16_t* __restrict__ Qb, const bf16_t* __restrict__ Kb,
    const bf16_t* __restrict__ VbT, bf16_t* __restrict__ Cc) {
  __shared__ __align__(16) bf16_t pt[8][16 * 32];

  const int tid = threadIdx.x, wave = tid >> 5, lane = tid & 31;
  const int bh = blockIdx.y, b = bh >> 3, h = bh & (HH - 1);
  const int q0 = (blockIdx.x << 7) + (wave << 4);

  const bf16_t* Qp = Qb + ((size_t)bh * SS + q0) * AA;
  v16bf qa0 = frag_a_k(Qp, AA);        // a = 0..31
  v16bf qa1 = frag_a_k(Qp + 32, AA);   // a = 32..63

  v8f acc0 = {}, acc1 = {}, acc2 = {}, acc3 = {};
  float mrow[8], lrow[8];
#pragma unroll
  for (int v = 0; v < 8; ++v) { mrow[v] = -1e30f; lrow[v] = 0.0f; }

  const bf16_t* Krow = Kb + (size_t)bh * SS * AA;   // [t][a]
  const bf16_t* Vtr  = VbT + (size_t)bh * AA * SS;  // [a][t]
  bf16_t* pp = pt[wave];

  // one 32-key step; consumes kc*, loads kn* for step jj+1 (clamped in-bounds)
  auto step = [&](int jj, v16bf& kc0, v16bf& kc1, v16bf& kc2, v16bf& kc3,
                  v16bf& kn0, v16bf& kn1, v16bf& kn2, v16bf& kn3) {
    const int t0 = jj << 5;

    // V fragments for this step FIRST (so PV waits leave K-prefetch in flight)
    v16bf vb0 = frag_b_k(Vtr + (size_t)(0 * 16) * SS + t0, SS);
    v16bf vb1 = frag_b_k(Vtr + (size_t)(1 * 16) * SS + t0, SS);
    v16bf vb2 = frag_b_k(Vtr + (size_t)(2 * 16) * SS + t0, SS);
    v16bf vb3 = frag_b_k(Vtr + (size_t)(3 * 16) * SS + t0, SS);

    // prefetch next step's K fragments (clamped address; harmless when unused)
    const int tn = (t0 + 64 <= SS) ? (t0 + 32) : t0;
    const bf16_t* kpn = Krow + (size_t)tn * AA;
    kn0 = frag_b_k(kpn, AA);
    kn1 = frag_b_k(kpn + 32, AA);
    kn2 = frag_b_k(kpn + 16 * AA, AA);
    kn3 = frag_b_k(kpn + 16 * AA + 32, AA);

    SCHED_FENCE();   // pin the load block above the compute

    // ---- scores: two 16x16 tiles over keys [t0, t0+32) ----
    v8f s0 = {}, s1 = {};
    s0 = WMMA_BF16(qa0, kc0, s0);
    s0 = WMMA_BF16(qa1, kc1, s0);
    s1 = WMMA_BF16(qa0, kc2, s1);
    s1 = WMMA_BF16(qa1, kc3, s1);

    SCHED_FENCE();

    // ---- online softmax over the 32 keys ----
    const int ncol = lane & 15;
    const int key0 = t0 + ncol, key1 = t0 + 16 + ncol;
    float p0[8], p1[8];
#pragma unroll
    for (int v = 0; v < 8; ++v) {
      const int qrow = q0 + ((lane >> 4) << 3) + v;
      float sv0 = s0[v] * 0.125f;            // 1/sqrt(64)
      float sv1 = s1[v] * 0.125f;
      if (key0 > qrow) sv0 = -1e30f;
      if (key1 > qrow) sv1 = -1e30f;
      float rmax = fmaxf(sv0, sv1);
#pragma unroll
      for (int off = 1; off < 16; off <<= 1)
        rmax = fmaxf(rmax, __shfl_xor(rmax, off, 32));
      const float mnew  = fmaxf(mrow[v], rmax);
      const float alpha = __expf(mrow[v] - mnew);
      const float pv0   = __expf(sv0 - mnew);
      const float pv1   = __expf(sv1 - mnew);
      float rsum = pv0 + pv1;
#pragma unroll
      for (int off = 1; off < 16; off <<= 1)
        rsum += __shfl_xor(rsum, off, 32);
      lrow[v] = lrow[v] * alpha + rsum;
      mrow[v] = mnew;
      p0[v] = pv0; p1[v] = pv1;
      acc0[v] *= alpha; acc1[v] *= alpha; acc2[v] *= alpha; acc3[v] *= alpha;
    }

    // ---- P (C-layout) -> LDS [m][32] -> A-fragment (K = 32, all real) ----
#pragma unroll
    for (int v = 0; v < 8; ++v) {
      const int m = ((lane >> 4) << 3) + v;
      pp[m * 32 + ncol]      = (bf16_t)p0[v];
      pp[m * 32 + 16 + ncol] = (bf16_t)p1[v];
    }
    __builtin_amdgcn_wave_barrier();   // same-wave LDS ops complete in order
    v16bf pa = frag_a_k(pp, 32);
    __builtin_amdgcn_wave_barrier();

    // ---- O += P @ V ; B[n=a][k=t] = VbT[a][t], t-contiguous ----
    acc0 = WMMA_BF16(pa, vb0, acc0);
    acc1 = WMMA_BF16(pa, vb1, acc1);
    acc2 = WMMA_BF16(pa, vb2, acc2);
    acc3 = WMMA_BF16(pa, vb3, acc3);
  };

  const int npair = (q0 >> 5) + 1;   // exact causal trip count
  v16bf kA0 = frag_b_k(Krow, AA);
  v16bf kA1 = frag_b_k(Krow + 32, AA);
  v16bf kA2 = frag_b_k(Krow + 16 * AA, AA);
  v16bf kA3 = frag_b_k(Krow + 16 * AA + 32, AA);
  v16bf kB0, kB1, kB2, kB3;

  int jj = 0;
  for (; jj + 2 <= npair; jj += 2) {
    step(jj,     kA0, kA1, kA2, kA3, kB0, kB1, kB2, kB3);
    step(jj + 1, kB0, kB1, kB2, kB3, kA0, kA1, kA2, kA3);
  }
  if (jj < npair)
    step(jj, kA0, kA1, kA2, kA3, kB0, kB1, kB2, kB3);

  // ---- finalize: divide by row sum, store head-major concat bf16 ----
  const int ncol = lane & 15;
#pragma unroll
  for (int v = 0; v < 8; ++v) {
    const int srow = q0 + ((lane >> 4) << 3) + v;
    const float inv = 1.0f / lrow[v];
    bf16_t* crow = Cc + ((size_t)b * SS + srow) * (HH * AA) + h * AA;
    crow[ 0 + ncol] = (bf16_t)(acc0[v] * inv);
    crow[16 + ncol] = (bf16_t)(acc1[v] * inv);
    crow[32 + ncol] = (bf16_t)(acc2[v] * inv);
    crow[48 + ncol] = (bf16_t)(acc3[v] * inv);
  }
}

// ---------------------------------------------------------------------------
// Output projection: out[r, a] = concat[r, :] @ wo[:, a] + bo[a], fp32 out.
// Wave owns a 16(r) x 64(a) tile; woT is [a][k]; double-buffered like QKV.
// ---------------------------------------------------------------------------
__global__ __launch_bounds__(128) void out_gemm_kernel(
    const bf16_t* __restrict__ Cc, const bf16_t* __restrict__ woT,
    const float* __restrict__ bo, float* __restrict__ out) {
  const int tid = threadIdx.x, wave = tid >> 5, lane = tid & 31;
  const int r0 = (blockIdx.x << 6) + (wave << 4);
  const int KD = HH * AA;   // 512

  const bf16_t* xrow = Cc + (size_t)r0 * KD;
  v8f acc0 = {}, acc1 = {}, acc2 = {}, acc3 = {};
  v16bf a_c  = frag_a_k(xrow, KD);
  v16bf b0_c = frag_b_k(woT + (size_t)(0 * 16) * KD, KD);
  v16bf b1_c = frag_b_k(woT + (size_t)(1 * 16) * KD, KD);
  v16bf b2_c = frag_b_k(woT + (size_t)(2 * 16) * KD, KD);
  v16bf b3_c = frag_b_k(woT + (size_t)(3 * 16) * KD, KD);
#pragma unroll
  for (int it = 0; it < KD / 32; ++it) {
    v16bf a_n, b0_n, b1_n, b2_n, b3_n;
    if (it + 1 < KD / 32) {
      const int k1 = (it + 1) * 32;
      a_n  = frag_a_k(xrow + k1, KD);
      b0_n = frag_b_k(woT + (size_t)(0 * 16) * KD + k1, KD);
      b1_n = frag_b_k(woT + (size_t)(1 * 16) * KD + k1, KD);
      b2_n = frag_b_k(woT + (size_t)(2 * 16) * KD + k1, KD);
      b3_n = frag_b_k(woT + (size_t)(3 * 16) * KD + k1, KD);
    }
    SCHED_FENCE();   // keep next-iter loads above this WMMA group
    acc0 = WMMA_BF16(a_c, b0_c, acc0);
    acc1 = WMMA_BF16(a_c, b1_c, acc1);
    acc2 = WMMA_BF16(a_c, b2_c, acc2);
    acc3 = WMMA_BF16(a_c, b3_c, acc3);
    SCHED_FENCE();
    if (it + 1 < KD / 32) {
      a_c = a_n; b0_c = b0_n; b1_c = b1_n; b2_c = b2_n; b3_c = b3_n;
    }
  }

  v8f* accs[4] = {&acc0, &acc1, &acc2, &acc3};
#pragma unroll
  for (int tn = 0; tn < 4; ++tn) {
    const int ncol = (tn << 4) + (lane & 15);
    const float bval = bo[ncol];
#pragma unroll
    for (int v = 0; v < 8; ++v) {
      const int m = ((lane >> 4) << 3) + v;
      out[(size_t)(r0 + m) * AA + ncol] = (*accs[tn])[v] + bval;
    }
  }
}

// ---------------------------------------------------------------------------
// Host launcher
// ---------------------------------------------------------------------------
extern "C" void kernel_launch(void* const* d_in, const int* in_sizes, int n_in,
                              void* d_out, int out_size, void* d_ws,
                              size_t ws_size, hipStream_t stream) {
  const float* x  = (const float*)d_in[0];
  // d_in[1] = causal mask (bool) -- structure known, unused
  const float* wq = (const float*)d_in[2];
  const float* bq = (const float*)d_in[3];
  const float* wk = (const float*)d_in[4];
  const float* bk = (const float*)d_in[5];
  const float* wv = (const float*)d_in[6];
  const float* bv = (const float*)d_in[7];
  const float* wo = (const float*)d_in[8];
  const float* bo = (const float*)d_in[9];
  float* out = (float*)d_out;

  // bf16 workspace carve-up (~21.6 MB, L2 resident)
  bf16_t* base = (bf16_t*)d_ws;
  size_t o = 0;
  bf16_t* xb  = base + o; o += (size_t)BB * SS * EE;
  bf16_t* wqT = base + o; o += (size_t)HH * AA * EE;
  bf16_t* wkT = base + o; o += (size_t)HH * AA * EE;
  bf16_t* wvT = base + o; o += (size_t)HH * AA * EE;
  bf16_t* woT = base + o; o += (size_t)AA * (HH * AA);
  bf16_t* Qb  = base + o; o += (size_t)BB * HH * SS * AA;
  bf16_t* Kb  = base + o; o += (size_t)BB * HH * SS * AA;
  bf16_t* VbT = base + o; o += (size_t)BB * HH * AA * SS;
  bf16_t* Cc  = base + o; o += (size_t)BB * SS * HH * AA;

  {
    int n8 = (BB * SS * EE) / 8;
    cvt_bf16x8_kernel<<<(n8 + 255) / 256, 256, 0, stream>>>(x, xb, n8);
    int n = HH * EE * AA;
    cvt_t_hea_kernel<<<(n + 255) / 256, 256, 0, stream>>>(wq, wqT);
    cvt_t_hea_kernel<<<(n + 255) / 256, 256, 0, stream>>>(wk, wkT);
    cvt_t_hea_kernel<<<(n + 255) / 256, 256, 0, stream>>>(wv, wvT);
    n = (HH * AA) * AA;
    cvt_t_wo_kernel<<<(n + 255) / 256, 256, 0, stream>>>(wo, woT);
  }

  // Q/K/V projections: grid (S/64, B*H, 3), 4 waves x 16x64 tiles
  qkv_gemm_kernel<<<dim3(SS / 64, BB * HH, 3), 128, 0, stream>>>(
      xb, wqT, wkT, wvT, bq, bk, bv, Qb, Kb, VbT);

  // causal flash attention: grid (S/128, B*H), 8 independent waves
  attn_kernel<<<dim3(SS / 128, BB * HH), 256, 0, stream>>>(Qb, Kb, VbT, Cc);

  // output projection: grid (B*S/64)
  out_gemm_kernel<<<dim3(BB * SS / 64), 128, 0, stream>>>(Cc, woT, bo, out);
}
